// Attention_37031208026188
// MI455X (gfx1250) — matrix operand order repaired and checked by reading, not compiled, not run on hardware
//
#include <hip/hip_runtime.h>

// Problem constants (fixed by the reference)
constexpr int NB = 8;
constexpr int NT = 2048;
constexpr int NS = 2048;
constexpr int ND = 1024;
constexpr size_t BTD = (size_t)NB * NT * ND;   // 16,777,216
constexpr size_t BSD = (size_t)NB * NS * ND;   // 16,777,216
constexpr size_t BTS = (size_t)NB * NT * NS;   // 33,554,432

typedef __attribute__((ext_vector_type(16))) __bf16 bf16x16;
typedef __attribute__((ext_vector_type(8)))  float  f32x8;

union FragU {
    bf16x16        v;
    uint4          q[2];
    unsigned short h[16];
};

__device__ __forceinline__ unsigned short f2bf(float f) {
    unsigned int u = __float_as_uint(f);
    u += 0x7FFFu + ((u >> 16) & 1u);          // round-to-nearest-even
    return (unsigned short)(u >> 16);
}

// ---------------------------------------------------------------------------
// Kernel 1: fp32 -> bf16 bulk convert (vectorized, n % 4 == 0)
// ---------------------------------------------------------------------------
__global__ __launch_bounds__(256) void cvt_f32_bf16(const float* __restrict__ in,
                                                    unsigned short* __restrict__ out,
                                                    size_t n) {
    size_t i = ((size_t)blockIdx.x * blockDim.x + threadIdx.x) * 4;
    if (i + 3 < n) {
        float4 f = *(const float4*)(in + i);
        ushort4 o;
        o.x = f2bf(f.x); o.y = f2bf(f.y); o.z = f2bf(f.z); o.w = f2bf(f.w);
        *(ushort4*)(out + i) = o;
    }
}

// ---------------------------------------------------------------------------
// Kernel 2: src (B,S,D) fp32 -> srcT (B,D,S) bf16, LDS-tiled transpose
// ---------------------------------------------------------------------------
__global__ __launch_bounds__(256) void transpose_bf16(const float* __restrict__ src,
                                                      unsigned short* __restrict__ dst) {
    __shared__ unsigned short tile[32][33];
    const int d0 = blockIdx.x * 32;
    const int s0 = blockIdx.y * 32;
    const float* sb = src + (size_t)blockIdx.z * NS * ND;
    unsigned short* db = dst + (size_t)blockIdx.z * ND * NS;
    const int tx = threadIdx.x;        // 0..31
    const int ty = threadIdx.y;        // 0..7
#pragma unroll
    for (int i = ty; i < 32; i += 8)
        tile[i][tx] = f2bf(sb[(size_t)(s0 + i) * ND + d0 + tx]);
    __syncthreads();
#pragma unroll
    for (int i = ty; i < 32; i += 8)
        db[(size_t)(d0 + i) * NS + s0 + tx] = tile[tx][i];
}

// ---------------------------------------------------------------------------
// Kernel 3: scores = trg . src^T  (bf16 WMMA, fp32 accum)
// grid = (NS/128, NT/128, NB), block = 128 (4 waves, each owns 64x64)
// ---------------------------------------------------------------------------
__global__ __launch_bounds__(128) void scores_wmma(const unsigned short* __restrict__ trgB,
                                                   const unsigned short* __restrict__ srcB,
                                                   float* __restrict__ outw) {
    const int lane = threadIdx.x & 31;
    const int wave = threadIdx.x >> 5;
    const int wm = wave >> 1, wn = wave & 1;
    const int b  = blockIdx.z;
    const int m0 = blockIdx.y * 128 + wm * 64;   // T rows
    const int n0 = blockIdx.x * 128 + wn * 64;   // S cols
    const int lh = lane & 15;
    const int hi = lane >> 4;

    const unsigned short* A  = trgB + (size_t)b * NT * ND;
    const unsigned short* Bm = srcB + (size_t)b * NS * ND;

    f32x8 c[4][4];
#pragma unroll
    for (int i = 0; i < 4; ++i)
#pragma unroll
        for (int j = 0; j < 4; ++j)
            c[i][j] = (f32x8){0.f, 0.f, 0.f, 0.f, 0.f, 0.f, 0.f, 0.f};

    const int aK = hi * 8;    // A-layout: lanes 0-15 -> K 0-7/16-23, lanes 16-31 -> K 8-15/24-31
    const int bK = hi * 8;    // B-layout: lanes 0-15 -> K 0-7/16-23, lanes 16-31 -> K 8-15/24-31

    for (int k0 = 0; k0 < ND; k0 += 32) {
        FragU a[4], bf[4];
#pragma unroll
        for (int i = 0; i < 4; ++i) {
            const unsigned short* p = A + (size_t)(m0 + 16 * i + lh) * ND + k0 + aK;
            a[i].q[0] = *(const uint4*)(p);
            a[i].q[1] = *(const uint4*)(p + 16);
        }
#pragma unroll
        for (int j = 0; j < 4; ++j) {
            const unsigned short* p = Bm + (size_t)(n0 + 16 * j + lh) * ND + k0 + bK;
            bf[j].q[0] = *(const uint4*)(p);
            bf[j].q[1] = *(const uint4*)(p + 16);
        }
#pragma unroll
        for (int i = 0; i < 4; ++i)
#pragma unroll
            for (int j = 0; j < 4; ++j)
                c[i][j] = __builtin_amdgcn_wmma_f32_16x16x32_bf16(
                    false, a[i].v, false, bf[j].v, (short)0, c[i][j], false, false);
    }

    float* O = outw + (size_t)b * NT * NS;
#pragma unroll
    for (int i = 0; i < 4; ++i)
#pragma unroll
        for (int j = 0; j < 4; ++j)
#pragma unroll
            for (int r = 0; r < 8; ++r) {
                const int row = m0 + 16 * i + r + hi * 8;
                const int col = n0 + 16 * j + lh;
                O[(size_t)row * NS + col] = c[i][j][r];
            }
}

// ---------------------------------------------------------------------------
// Kernel 4: in-place row softmax over S (one block per (b,t) row)
// ---------------------------------------------------------------------------
__global__ __launch_bounds__(256) void softmax_rows(float* __restrict__ w) {
    __shared__ float red[256];
    float* p = w + (size_t)blockIdx.x * NS;
    const int tid = threadIdx.x;

    float x[8];
    float m = -INFINITY;
#pragma unroll
    for (int i = 0; i < 8; ++i) { x[i] = p[tid + i * 256]; m = fmaxf(m, x[i]); }
    red[tid] = m; __syncthreads();
    for (int off = 128; off > 0; off >>= 1) {
        if (tid < off) red[tid] = fmaxf(red[tid], red[tid + off]);
        __syncthreads();
    }
    m = red[0]; __syncthreads();

    float s = 0.f;
#pragma unroll
    for (int i = 0; i < 8; ++i) { x[i] = __expf(x[i] - m); s += x[i]; }
    red[tid] = s; __syncthreads();
    for (int off = 128; off > 0; off >>= 1) {
        if (tid < off) red[tid] += red[tid + off];
        __syncthreads();
    }
    const float inv = 1.0f / red[0];
#pragma unroll
    for (int i = 0; i < 8; ++i) p[tid + i * 256] = x[i] * inv;
}

// ---------------------------------------------------------------------------
// Kernel 5: context = W . src  (A = fp32 weights cvt'd on the fly, B = srcT bf16)
// grid = (ND/128, NT/128, NB), block = 128
// ---------------------------------------------------------------------------
__global__ __launch_bounds__(128) void context_wmma(const float* __restrict__ Wf,
                                                    const unsigned short* __restrict__ srcT,
                                                    float* __restrict__ ctx) {
    const int lane = threadIdx.x & 31;
    const int wave = threadIdx.x >> 5;
    const int wm = wave >> 1, wn = wave & 1;
    const int b  = blockIdx.z;
    const int m0 = blockIdx.y * 128 + wm * 64;   // T rows
    const int n0 = blockIdx.x * 128 + wn * 64;   // D cols
    const int lh = lane & 15;
    const int hi = lane >> 4;

    const float*          W  = Wf   + (size_t)b * NT * NS;
    const unsigned short* Bt = srcT + (size_t)b * ND * NS;

    f32x8 c[4][4];
#pragma unroll
    for (int i = 0; i < 4; ++i)
#pragma unroll
        for (int j = 0; j < 4; ++j)
            c[i][j] = (f32x8){0.f, 0.f, 0.f, 0.f, 0.f, 0.f, 0.f, 0.f};

    const int aK = hi * 8;
    const int bK = hi * 8;

    for (int k0 = 0; k0 < NS; k0 += 32) {
        FragU a[4], bf[4];
#pragma unroll
        for (int i = 0; i < 4; ++i) {
            const float* p = W + (size_t)(m0 + 16 * i + lh) * NS + k0 + aK;
            float4 f0 = *(const float4*)(p);
            float4 f1 = *(const float4*)(p + 4);
            float4 f2 = *(const float4*)(p + 16);
            float4 f3 = *(const float4*)(p + 20);
            a[i].h[0]  = f2bf(f0.x); a[i].h[1]  = f2bf(f0.y);
            a[i].h[2]  = f2bf(f0.z); a[i].h[3]  = f2bf(f0.w);
            a[i].h[4]  = f2bf(f1.x); a[i].h[5]  = f2bf(f1.y);
            a[i].h[6]  = f2bf(f1.z); a[i].h[7]  = f2bf(f1.w);
            a[i].h[8]  = f2bf(f2.x); a[i].h[9]  = f2bf(f2.y);
            a[i].h[10] = f2bf(f2.z); a[i].h[11] = f2bf(f2.w);
            a[i].h[12] = f2bf(f3.x); a[i].h[13] = f2bf(f3.y);
            a[i].h[14] = f2bf(f3.z); a[i].h[15] = f2bf(f3.w);
        }
#pragma unroll
        for (int j = 0; j < 4; ++j) {
            const unsigned short* p = Bt + (size_t)(n0 + 16 * j + lh) * NS + k0 + bK;
            bf[j].q[0] = *(const uint4*)(p);
            bf[j].q[1] = *(const uint4*)(p + 16);
        }
#pragma unroll
        for (int i = 0; i < 4; ++i)
#pragma unroll
            for (int j = 0; j < 4; ++j)
                c[i][j] = __builtin_amdgcn_wmma_f32_16x16x32_bf16(
                    false, a[i].v, false, bf[j].v, (short)0, c[i][j], false, false);
    }

    float* O = ctx + (size_t)b * NT * ND;
#pragma unroll
    for (int i = 0; i < 4; ++i)
#pragma unroll
        for (int j = 0; j < 4; ++j)
#pragma unroll
            for (int r = 0; r < 8; ++r) {
                const int row = m0 + 16 * i + r + hi * 8;
                const int col = n0 + 16 * j + lh;
                O[(size_t)row * ND + col] = c[i][j][r];
            }
}

// ---------------------------------------------------------------------------
extern "C" void kernel_launch(void* const* d_in, const int* in_sizes, int n_in,
                              void* d_out, int out_size, void* d_ws, size_t ws_size,
                              hipStream_t stream) {
    const float* trg = (const float*)d_in[0];   // (B,T,D) fp32
    const float* src = (const float*)d_in[1];   // (B,S,D) fp32

    float* ctx = (float*)d_out;                 // (B,T,D) context first
    float* wts = (float*)d_out + BTD;           // (B,T,S) attention weights

    // workspace: bf16 copies (3 * 32 MB = 96 MB)
    unsigned short* trgB = (unsigned short*)d_ws;
    unsigned short* srcB = trgB + BTD;
    unsigned short* srcT = srcB + BSD;

    // 1) fp32 -> bf16 converts
    cvt_f32_bf16<<<(unsigned)(BTD / 4 / 256), 256, 0, stream>>>(trg, trgB, BTD);
    cvt_f32_bf16<<<(unsigned)(BSD / 4 / 256), 256, 0, stream>>>(src, srcB, BSD);

    // 2) src -> srcT (B,D,S) bf16
    transpose_bf16<<<dim3(ND / 32, NS / 32, NB), dim3(32, 8), 0, stream>>>(src, srcT);

    // 3) raw scores into the weights output region
    scores_wmma<<<dim3(NS / 128, NT / 128, NB), 128, 0, stream>>>(trgB, srcB, wts);

    // 4) in-place softmax over S
    softmax_rows<<<(unsigned)(NB * NT), 256, 0, stream>>>(wts);

    // 5) context = weights . src
    context_wmma<<<dim3(ND / 128, NT / 128, NB), 128, 0, stream>>>(wts, srcT, ctx);
}
